// AttnDecoderRNN_20950850470320
// MI455X (gfx1250) — matrix operand order repaired and checked by reading, not compiled, not run on hardware
//
#include <hip/hip_runtime.h>
#include <hip/hip_bf16.h>
#include <math.h>

typedef __bf16 bf16;
typedef bf16 v16bf __attribute__((ext_vector_type(16)));
typedef bf16 v8bf  __attribute__((ext_vector_type(8)));
typedef bf16 v4bf  __attribute__((ext_vector_type(4)));
typedef float v8f  __attribute__((ext_vector_type(8)));

#define HH 128
#define WW 512
#define P_TOTAL 65536         // H*W
#define CCH 684               // channels
#define NPOS 64               // positions per block in k_attn
#define NBLK (P_TOTAL / NPOS) // 1024
#define KCHUNKS 22            // ceil(684/32); chunks 0..20 are full, 21 is the tail

__device__ __forceinline__ float sigmoidf_(float x) { return 1.0f / (1.0f + expf(-x)); }

// ---------------------------------------------------------------------------
// K1: embedding lookup + GRU1 cell + hidden1
// ---------------------------------------------------------------------------
__global__ __launch_bounds__(256) void k_setup(
    const int* __restrict__ input, const float* __restrict__ hidden,
    const float* __restrict__ emb_table,
    const float* __restrict__ Wih, const float* __restrict__ Whh,
    const float* __restrict__ bih, const float* __restrict__ bhh,
    const float* __restrict__ W_hidden, const float* __restrict__ b_hidden,
    float* __restrict__ ws_emb, float* __restrict__ ws_st, float* __restrict__ ws_h1)
{
    __shared__ float emb_s[256], gi[768], gh[768], st_s[256];
    const int t = threadIdx.x;
    const int tok = input[0];
    emb_s[t] = emb_table[tok * 256 + t];
    __syncthreads();
    for (int r = t; r < 768; r += 256) {
        float a = bih[r], b = bhh[r];
        const float* wi = Wih + r * 256;
        const float* wh = Whh + r * 256;
        for (int k = 0; k < 256; ++k) { a += emb_s[k] * wi[k]; b += hidden[k] * wh[k]; }
        gi[r] = a; gh[r] = b;
    }
    __syncthreads();
    {
        float r = sigmoidf_(gi[t] + gh[t]);
        float z = sigmoidf_(gi[256 + t] + gh[256 + t]);
        float n = tanhf(gi[512 + t] + r * gh[512 + t]);
        float st = (1.0f - z) * n + z * hidden[t];
        st_s[t] = st; ws_st[t] = st; ws_emb[t] = emb_s[t];
    }
    __syncthreads();
    {
        float h1 = b_hidden[t];
        const float* wrow = W_hidden + t * 256;
        for (int k = 0; k < 256; ++k) h1 += st_s[k] * wrow[k];
        ws_h1[t] = h1;
    }
}

// ---------------------------------------------------------------------------
// K2: 3x3 SAME conv on decoder_attention + attention_sum  -> d_out attn_sum
// ---------------------------------------------------------------------------
__global__ __launch_bounds__(256) void k_conv(
    const float* __restrict__ attn_sum_in, const float* __restrict__ dec_attn,
    const float* __restrict__ conv_w, const float* __restrict__ conv_b,
    float* __restrict__ out_asum)
{
    const int p = blockIdx.x * 256 + threadIdx.x;
    const int h = p >> 9, w = p & 511;
    float acc = conv_b[0];
#pragma unroll
    for (int dy = -1; dy <= 1; ++dy)
#pragma unroll
        for (int dx = -1; dx <= 1; ++dx) {
            int hh = h + dy, ww = w + dx;
            if (hh >= 0 && hh < HH && ww >= 0 && ww < WW)
                acc += conv_w[(dy + 1) * 3 + (dx + 1)] * dec_attn[hh * WW + ww];
        }
    out_asum[p] = attn_sum_in[p] + acc;
}

// ---------------------------------------------------------------------------
// K3: fused attention-energy GEMM (bf16 WMMA) + tanh + W_v dot + exp
//     Each block: M=256 x N=64 tile, K=684 (padded 704) in chunks of 32.
//     Double-buffered LDS, software-pipelined global->LDS staging.
//     8 waves; wave w owns M-tiles {2w,2w+1} x N-tiles {0..3}.
// ---------------------------------------------------------------------------
__global__ __launch_bounds__(256) void k_attn(
    const float* __restrict__ enc,     // (684, 65536)
    const float* __restrict__ W_ua,    // (256, 684)
    const float* __restrict__ b_ua,    // (256)
    const float* __restrict__ W_uf,    // (256)
    const float* __restrict__ b_uf,    // (256)
    const float* __restrict__ W_v,     // (256)
    const float* __restrict__ b_v,     // (1)
    const float* __restrict__ h1,      // (256)  hidden1
    const float* __restrict__ asum,    // (65536) attn_sum (from d_out)
    float* __restrict__ escore,        // (65536) unnormalized exp scores
    float* __restrict__ Spart)         // (NBLK)
{
    __shared__ bf16 As[2][256][40];    // A chunk: [buf][m][k], 80B row stride
    __shared__ bf16 Bs[2][64][40];     // B chunk: [buf][p][k]
    __shared__ float scorePart[16][64];
    __shared__ float esLds[64];

    const int tid  = threadIdx.x;
    const int wv   = tid >> 5;
    const int lane = tid & 31;
    const int half = lane >> 4;        // 0: lanes 0-15, 1: lanes 16-31
    const int l16  = lane & 15;
    const int p0   = blockIdx.x * NPOS;

    v8f zero8 = {0.f,0.f,0.f,0.f,0.f,0.f,0.f,0.f};
    v8f acc[2][4];
#pragma unroll
    for (int mi = 0; mi < 2; ++mi)
#pragma unroll
        for (int nt = 0; nt < 4; ++nt) acc[mi][nt] = zero8;

    float4 ra[8];   // prefetch regs: A = 8 x float4 per thread (256x32 elems)
    float  rb[8];   // prefetch regs: B = 8 floats per thread   (32x64 elems)

    // issue global loads for chunk kc into registers (guard-free for full chunks)
    auto load_chunk = [&](int kc) {
        const int c0 = kc * 32;
        if (c0 + 32 <= CCH) {
#pragma unroll
            for (int it = 0; it < 8; ++it) {
                int idx = it * 1024 + tid * 4;
                int m = idx >> 5, j = idx & 31;
                ra[it] = *reinterpret_cast<const float4*>(W_ua + m * CCH + c0 + j);
            }
#pragma unroll
            for (int it = 0; it < 2; ++it) {
                int i = it * 256 + tid;
                int p = i & 63, j4 = (i >> 6) << 2;
                const float* b = enc + (size_t)(c0 + j4) * P_TOTAL + p0 + p;
#pragma unroll
                for (int d = 0; d < 4; ++d) rb[it * 4 + d] = b[(size_t)d * P_TOTAL];
            }
        } else {  // tail chunk (c = 672..683 valid, rest zero) -- runs once
#pragma unroll
            for (int it = 0; it < 8; ++it) {
                int idx = it * 1024 + tid * 4;
                int m = idx >> 5, j = idx & 31;
                const float* w = W_ua + m * CCH;
                float4 f;
                f.x = (c0 + j + 0 < CCH) ? w[c0 + j + 0] : 0.0f;
                f.y = (c0 + j + 1 < CCH) ? w[c0 + j + 1] : 0.0f;
                f.z = (c0 + j + 2 < CCH) ? w[c0 + j + 2] : 0.0f;
                f.w = (c0 + j + 3 < CCH) ? w[c0 + j + 3] : 0.0f;
                ra[it] = f;
            }
#pragma unroll
            for (int it = 0; it < 2; ++it) {
                int i = it * 256 + tid;
                int p = i & 63, j4 = (i >> 6) << 2;
#pragma unroll
                for (int d = 0; d < 4; ++d) {
                    int c = c0 + j4 + d;
                    rb[it * 4 + d] = (c < CCH) ? enc[(size_t)c * P_TOTAL + p0 + p] : 0.0f;
                }
            }
        }
    };

    // convert + store staged registers into LDS buffer `buf`
    auto store_chunk = [&](int buf) {
#pragma unroll
        for (int it = 0; it < 8; ++it) {
            int idx = it * 1024 + tid * 4;
            int m = idx >> 5, j = idx & 31;
            v4bf v = { (bf16)ra[it].x, (bf16)ra[it].y, (bf16)ra[it].z, (bf16)ra[it].w };
            *reinterpret_cast<v4bf*>(&As[buf][m][j]) = v;
        }
#pragma unroll
        for (int it = 0; it < 2; ++it) {
            int i = it * 256 + tid;
            int p = i & 63, j4 = (i >> 6) << 2;
            v4bf v = { (bf16)rb[it*4+0], (bf16)rb[it*4+1], (bf16)rb[it*4+2], (bf16)rb[it*4+3] };
            *reinterpret_cast<v4bf*>(&Bs[buf][p][j4]) = v;
        }
    };

    // fragment assembly + 8 WMMAs from LDS buffer `buf`
    auto compute = [&](int buf) {
        const int kb = half * 16;
        v16bf bfrag[4];
#pragma unroll
        for (int nt = 0; nt < 4; ++nt) {
            const bf16* src = &Bs[buf][nt * 16 + l16][kb];
            v8bf b0 = *reinterpret_cast<const v8bf*>(src);
            v8bf b1 = *reinterpret_cast<const v8bf*>(src + 8);
            bfrag[nt] = __builtin_shufflevector(b0, b1, 0,1,2,3,4,5,6,7,8,9,10,11,12,13,14,15);
        }
        const int base = half * 16;
#pragma unroll
        for (int mi = 0; mi < 2; ++mi) {
            int m = (2 * wv + mi) * 16 + l16;
            v8bf a0 = *reinterpret_cast<const v8bf*>(&As[buf][m][base]);
            v8bf a1 = *reinterpret_cast<const v8bf*>(&As[buf][m][base + 8]);
            v16bf afrag = __builtin_shufflevector(a0, a1, 0,1,2,3,4,5,6,7,8,9,10,11,12,13,14,15);
#pragma unroll
            for (int nt = 0; nt < 4; ++nt)
                acc[mi][nt] = __builtin_amdgcn_wmma_f32_16x16x32_bf16(
                    false, afrag, false, bfrag[nt], (short)0, acc[mi][nt], false, false);
        }
    };

    // prologue: stage chunk 0
    load_chunk(0);
    store_chunk(0);
    __syncthreads();

    for (int kc = 0; kc < KCHUNKS; ++kc) {
        if (kc + 1 < KCHUNKS) load_chunk(kc + 1);   // global loads in flight
        compute(kc & 1);                            // WMMAs overlap the loads
        if (kc + 1 < KCHUNKS) store_chunk((kc + 1) & 1);
        __syncthreads();
    }

    // ---- epilogue: et = tanh(acc + h1[m] + b_ua[m] + asum[p]*W_uf[m] + b_uf[m])
    //               score[p] += W_v[m] * et  (deterministic two-stage reduce)
    float hv[2][8], wuf[2][8], wvv[2][8];
#pragma unroll
    for (int mi = 0; mi < 2; ++mi)
#pragma unroll
        for (int v = 0; v < 8; ++v) {
            int m = (2 * wv + mi) * 16 + v + 8 * half;
            hv[mi][v]  = h1[m] + b_ua[m] + b_uf[m];
            wuf[mi][v] = W_uf[m];
            wvv[mi][v] = W_v[m];
        }
    float asl[4];
#pragma unroll
    for (int nt = 0; nt < 4; ++nt) asl[nt] = asum[p0 + nt * 16 + l16];

    float sp[4] = {0.f, 0.f, 0.f, 0.f};
#pragma unroll
    for (int mi = 0; mi < 2; ++mi)
#pragma unroll
        for (int nt = 0; nt < 4; ++nt)
#pragma unroll
            for (int v = 0; v < 8; ++v) {
                float x = acc[mi][nt][v] + hv[mi][v] + asl[nt] * wuf[mi][v];
                sp[nt] += wvv[mi][v] * tanhf(x);
            }
#pragma unroll
    for (int nt = 0; nt < 4; ++nt)
        scorePart[2 * wv + half][nt * 16 + l16] = sp[nt];
    __syncthreads();

    if (tid < 64) {
        float s = b_v[0];
#pragma unroll
        for (int h2 = 0; h2 < 16; ++h2) s += scorePart[h2][tid];
        float es = expf(s);            // no max-shift, matching the reference
        escore[p0 + tid] = es;
        esLds[tid] = es;
    }
    __syncthreads();
    if (tid == 0) {
        float S = 0.f;
        for (int t = 0; t < 64; ++t) S += esLds[t];
        Spart[blockIdx.x] = S;
    }
}

// ---------------------------------------------------------------------------
// K4: deterministic softmax-denominator reduction
// ---------------------------------------------------------------------------
__global__ __launch_bounds__(256) void k_sreduce(const float* __restrict__ Spart,
                                                 float* __restrict__ S)
{
    __shared__ float l[256];
    float s = 0.f;
    for (int i = threadIdx.x; i < NBLK; i += 256) s += Spart[i];
    l[threadIdx.x] = s; __syncthreads();
    for (int st = 128; st > 0; st >>= 1) {
        if (threadIdx.x < st) l[threadIdx.x] += l[threadIdx.x + st];
        __syncthreads();
    }
    if (threadIdx.x == 0) S[0] = l[0];
}

// ---------------------------------------------------------------------------
// K5: ct[c] = sum_p escore[p]*enc[c,p] / S   (one block per channel)
// ---------------------------------------------------------------------------
__global__ __launch_bounds__(256) void k_ct(const float* __restrict__ enc,
                                            const float* __restrict__ escore,
                                            const float* __restrict__ S,
                                            float* __restrict__ ct)
{
    __shared__ float l[256];
    const int c = blockIdx.x;
    const float4* e4 = reinterpret_cast<const float4*>(enc + (size_t)c * P_TOTAL);
    const float4* s4 = reinterpret_cast<const float4*>(escore);
    float acc = 0.f;
    for (int i = threadIdx.x; i < P_TOTAL / 4; i += 256) {
        float4 a = e4[i]; float4 b = s4[i];
        acc += a.x * b.x + a.y * b.y + a.z * b.z + a.w * b.w;
    }
    l[threadIdx.x] = acc; __syncthreads();
    for (int st = 128; st > 0; st >>= 1) {
        if (threadIdx.x < st) l[threadIdx.x] += l[threadIdx.x + st];
        __syncthreads();
    }
    if (threadIdx.x == 0) ct[c] = l[0] / S[0];
}

// ---------------------------------------------------------------------------
// K6: et_div output = escore / S
// ---------------------------------------------------------------------------
__global__ __launch_bounds__(256) void k_etdiv(const float* __restrict__ escore,
                                               const float* __restrict__ S,
                                               float* __restrict__ out_etdiv)
{
    const int p = blockIdx.x * 256 + threadIdx.x;
    out_etdiv[p] = escore[p] / S[0];
}

// ---------------------------------------------------------------------------
// K7: GRU cell #2 + output head + log_softmax
// ---------------------------------------------------------------------------
__global__ __launch_bounds__(256) void k_final(
    const float* __restrict__ ws_ct, const float* __restrict__ ws_st,
    const float* __restrict__ ws_emb,
    const float* __restrict__ gWih, const float* __restrict__ gWhh,
    const float* __restrict__ gbih, const float* __restrict__ gbhh,
    const float* __restrict__ W_h2, const float* __restrict__ b_h2,
    const float* __restrict__ W_e2, const float* __restrict__ b_e2,
    const float* __restrict__ W_wc, const float* __restrict__ b_wc,
    const float* __restrict__ W_out, const float* __restrict__ b_out,
    float* __restrict__ out_logits, float* __restrict__ out_h)
{
    __shared__ float ct_s[CCH], gi[768], gh[768], hn_s[256], pre[128], logits[128], red[2];
    const int t = threadIdx.x;
    for (int i = t; i < CCH; i += 256) ct_s[i] = ws_ct[i];
    __syncthreads();
    for (int r = t; r < 768; r += 256) {
        float a = gbih[r], b = gbhh[r];
        const float* wi = gWih + r * CCH;
        const float* wh = gWhh + r * 256;
        for (int k = 0; k < CCH; ++k) a += ct_s[k] * wi[k];
        for (int k = 0; k < 256; ++k) b += ws_st[k] * wh[k];
        gi[r] = a; gh[r] = b;
    }
    __syncthreads();
    {
        float r = sigmoidf_(gi[t] + gh[t]);
        float z = sigmoidf_(gi[256 + t] + gh[256 + t]);
        float n = tanhf(gi[512 + t] + r * gh[512 + t]);
        float h = (1.0f - z) * n + z * ws_st[t];
        hn_s[t] = h; out_h[t] = h;
    }
    __syncthreads();
    if (t < 128) {
        float h2 = b_h2[t], e2 = b_e2[t], c2 = b_wc[t];
        const float* w1 = W_h2 + t * 256;
        const float* w2 = W_e2 + t * 256;
        const float* w3 = W_wc + t * CCH;
        for (int k = 0; k < 256; ++k) { h2 += hn_s[k] * w1[k]; e2 += ws_emb[k] * w2[k]; }
        for (int k = 0; k < CCH; ++k) c2 += ct_s[k] * w3[k];
        pre[t] = h2 + e2 + c2;
    }
    __syncthreads();
    if (t < 128) {
        float l = b_out[t];
        const float* w = W_out + t * 128;
        for (int k = 0; k < 128; ++k) l += pre[k] * w[k];
        logits[t] = l;
    }
    __syncthreads();
    if (t == 0) {
        float m = logits[0];
        for (int k = 1; k < 128; ++k) m = fmaxf(m, logits[k]);
        float s = 0.f;
        for (int k = 0; k < 128; ++k) s += expf(logits[k] - m);
        red[0] = m; red[1] = logf(s);
    }
    __syncthreads();
    if (t < 128) out_logits[t] = logits[t] - red[0] - red[1];
}

// ---------------------------------------------------------------------------
extern "C" void kernel_launch(void* const* d_in, const int* in_sizes, int n_in,
                              void* d_out, int out_size, void* d_ws, size_t ws_size,
                              hipStream_t stream)
{
    // inputs (setup_inputs order)
    const int*   input     = (const int*)  d_in[0];
    const float* hidden    = (const float*)d_in[1];
    const float* enc       = (const float*)d_in[2];
    // d_in[3] = bb (unused), d_in[6] = dense_input (unused)
    const float* attn_sum0 = (const float*)d_in[4];
    const float* dec_attn  = (const float*)d_in[5];
    const float* emb_table = (const float*)d_in[7];
    const float* gru_Wih   = (const float*)d_in[8];
    const float* gru_Whh   = (const float*)d_in[9];
    const float* gru_bih   = (const float*)d_in[10];
    const float* gru_bhh   = (const float*)d_in[11];
    const float* gru1_Wih  = (const float*)d_in[12];
    const float* gru1_Whh  = (const float*)d_in[13];
    const float* gru1_bih  = (const float*)d_in[14];
    const float* gru1_bhh  = (const float*)d_in[15];
    const float* W_out     = (const float*)d_in[16];
    const float* b_out     = (const float*)d_in[17];
    const float* W_hidden  = (const float*)d_in[18];
    const float* b_hidden  = (const float*)d_in[19];
    const float* conv_w    = (const float*)d_in[20];
    const float* conv_b    = (const float*)d_in[21];
    const float* W_h2      = (const float*)d_in[22];
    const float* b_h2      = (const float*)d_in[23];
    const float* W_e2      = (const float*)d_in[24];
    const float* b_e2      = (const float*)d_in[25];
    const float* W_ua      = (const float*)d_in[26];
    const float* b_ua      = (const float*)d_in[27];
    const float* W_uf      = (const float*)d_in[28];
    const float* b_uf      = (const float*)d_in[29];
    const float* W_v       = (const float*)d_in[30];
    const float* b_v       = (const float*)d_in[31];
    const float* W_wc      = (const float*)d_in[32];
    const float* b_wc      = (const float*)d_in[33];

    // outputs: [logits 128][h_next 256][et_div 65536][attn_sum 65536]
    float* out        = (float*)d_out;
    float* out_logits = out;
    float* out_h      = out + 128;
    float* out_etdiv  = out + 384;
    float* out_asum   = out + 384 + P_TOTAL;

    // workspace layout (floats)
    float* W       = (float*)d_ws;
    float* escore  = W;                      // 65536
    float* Spart   = W + P_TOTAL;            // 1024
    float* Ssum    = W + P_TOTAL + NBLK;     // 1
    float* ws_emb  = Ssum + 4;               // 256
    float* ws_st   = ws_emb + 256;           // 256
    float* ws_h1   = ws_st + 256;            // 256
    float* ws_ct   = ws_h1 + 256;            // 684

    k_setup<<<1, 256, 0, stream>>>(input, hidden, emb_table,
                                   gru1_Wih, gru1_Whh, gru1_bih, gru1_bhh,
                                   W_hidden, b_hidden, ws_emb, ws_st, ws_h1);
    k_conv<<<P_TOTAL / 256, 256, 0, stream>>>(attn_sum0, dec_attn, conv_w, conv_b, out_asum);
    k_attn<<<NBLK, 256, 0, stream>>>(enc, W_ua, b_ua, W_uf, b_uf, W_v, b_v,
                                     ws_h1, out_asum, escore, Spart);
    k_sreduce<<<1, 256, 0, stream>>>(Spart, Ssum);
    k_ct<<<CCH, 256, 0, stream>>>(enc, escore, Ssum, ws_ct);
    k_etdiv<<<P_TOTAL / 256, 256, 0, stream>>>(escore, Ssum, out_etdiv);
    k_final<<<1, 256, 0, stream>>>(ws_ct, ws_st, ws_emb,
                                   gru_Wih, gru_Whh, gru_bih, gru_bhh,
                                   W_h2, b_h2, W_e2, b_e2, W_wc, b_wc,
                                   W_out, b_out, out_logits, out_h);
}
